// DeformableAlignmentFusion_16810501997304
// MI455X (gfx1250) — matrix operand order repaired and checked by reading, not compiled, not run on hardware
//
#include <hip/hip_runtime.h>
#include <cstdint>

// ---------------------------------------------------------------------------
// DeformableAlignmentFusion on gfx1250 (MI455X).
// All three convolutions are implicit GEMMs on v_wmma_f32_16x16x32_f16.
// Tap-major K ordering (no divisions in hot loops); A fragments staged in LDS
// as contiguous 8-half runs (ds_load_b128). ALL weight (B) streams are moved
// into LDS by the Tensor Data Mover, double buffered on TENSORcnt.
// Epilogues are fully vectorized (b128 loads/stores, no divergence).
// ---------------------------------------------------------------------------

typedef __attribute__((ext_vector_type(16))) _Float16 v16h;
typedef __attribute__((ext_vector_type(8)))  _Float16 v8h;
typedef __attribute__((ext_vector_type(8)))  float    v8f;
typedef __attribute__((ext_vector_type(4)))  float    v4f;

constexpr int Bn = 4, Cch = 128, Hh = 128, Wd = 128;
constexpr int HW = Hh * Wd;
constexpr int KOFF = 2304;   // 9 taps * 256 ch    (k = tap*256 + cin)
constexpr int KDEF = 1152;   // 9 taps * 128 ch    (k = tap*128 + cin)
constexpr int KMOD = 256;    // 256 ch             (k = cin)
constexpr int XT  = Wd / 16;

constexpr int WOFF_HALVES = (KOFF / 32) * 2 * 32 * 16;  // 72 kb * 2 nt
constexpr int WMOD_HALVES = (KMOD / 32) * 8 * 32 * 16;  //  8 kb * 8 nt
constexpr int WDEF_HALVES = (KDEF / 32) * 8 * 32 * 16;  // 36 kb * 8 nt

// ---------------------------------------------------------------------------
// Tensor Data Mover support (guarded; falls back to plain global loads).
// ---------------------------------------------------------------------------
#if defined(__AMDGCN__) && __has_builtin(__builtin_amdgcn_tensor_load_to_lds) && \
    __has_builtin(__builtin_amdgcn_s_wait_tensorcnt)
#define HAVE_TDM 1
#else
#define HAVE_TDM 0
#endif

#if HAVE_TDM
typedef __attribute__((ext_vector_type(4))) unsigned int tdm_v4u;
typedef __attribute__((ext_vector_type(8))) int          tdm_v8i;
typedef __attribute__((ext_vector_type(4))) int          tdm_v4i;

// 1-D TDM load: ndwords consecutive dwords from gsrc -> LDS byte offset lds_off.
__device__ __forceinline__ void tdm_load_1d(unsigned lds_off, const void* gsrc, int ndwords)
{
    unsigned long long ga = (unsigned long long)(uintptr_t)gsrc;
    tdm_v4u g0;
    g0[0] = 1u;                                               // count=1 (valid)
    g0[1] = lds_off;                                          // LDS byte address
    g0[2] = (unsigned)(ga & 0xFFFFFFFFu);                     // global_addr[31:0]
    g0[3] = (unsigned)((ga >> 32) & 0x1FFFFFFu) | (2u << 30); // addr[56:32] | type=2
    tdm_v8i g1;
    g1[0] = (int)(2u << 16);                                  // data_size = 2 (4 bytes)
    g1[1] = (int)(((unsigned)ndwords & 0xFFFFu) << 16);       // tensor_dim0[15:0]
    g1[2] = (int)((((unsigned)ndwords >> 16) & 0xFFFFu) | (1u << 16)); // dim0 hi | tensor_dim1=1
    g1[3] = (int)(((unsigned)ndwords & 0xFFFFu) << 16);       // tile_dim0
    g1[4] = 1;                                                // tile_dim1=1, tile_dim2=0
    g1[5] = ndwords;                                          // tensor_dim0_stride[31:0]
    g1[6] = 0;
    g1[7] = 0;
    tdm_v4i z4 = (tdm_v4i)0;
#if defined(__clang_major__) && (__clang_major__ >= 23)
    tdm_v8i z8 = (tdm_v8i)0;
    __builtin_amdgcn_tensor_load_to_lds(g0, g1, z4, z4, z8, 0);
#else
    __builtin_amdgcn_tensor_load_to_lds(g0, g1, z4, z4, 0);
#endif
}
#endif // HAVE_TDM

// ---------------------------------------------------------------------------
// Pack f32 weights into the WMMA B register image, tap-major K.
// ---------------------------------------------------------------------------
__global__ void pack_b_kernel(const float* __restrict__ src, _Float16* __restrict__ dst,
                              int Kdim, int Ntiles, int Nused, int taps, int cins_log2)
{
    int idx = blockIdx.x * 256 + threadIdx.x;
    int total = (Kdim / 32) * Ntiles * 32 * 16;
    if (idx >= total) return;
    int j    = idx & 15;
    int lane = (idx >> 4) & 31;
    int nt   = (idx >> 9) % Ntiles;
    int kb   = (idx >> 9) / Ntiles;
    int k = kb * 32 + (lane >> 4) * 8 + (j & 7) + (j >> 3) * 16;
    int n = nt * 16 + (lane & 15);
    int tap = k >> cins_log2;
    int cin = k & ((1 << cins_log2) - 1);
    float v = (n < Nused) ? src[(size_t)n * Kdim + (size_t)cin * taps + tap] : 0.0f;
    dst[idx] = (_Float16)v;
}

// A fragment: halves j=0..7 at p[0..7], j=8..15 at p[16..23] (ISA 16-bit A layout).
__device__ __forceinline__ v16h load_a_frag(const _Float16* p)
{
    v8h lo = *(const v8h*)p;
    v8h hi = *(const v8h*)(p + 16);
    v16h a;
#pragma unroll
    for (int j = 0; j < 8; ++j) { a[j] = lo[j]; a[j + 8] = hi[j]; }
    return a;
}

// ---------------------------------------------------------------------------
// Offset conv: 18ch = conv3x3(cat(low,high)) + b_off.
// One wave per (b, y, 16-px tile). M=16, N=32(18 used), K=2304 tap-major.
// ---------------------------------------------------------------------------
__global__ __launch_bounds__(32) void offset_conv_kernel(
    const float* __restrict__ low, const float* __restrict__ high,
    const _Float16* __restrict__ wB, const float* __restrict__ b_off,
    float* __restrict__ offs)
{
    constexpr int PS = 264;                          // padded cin stride (halves)
    __shared__ __align__(16) _Float16 patch[3 * 18 * PS];
    __shared__ __align__(16) _Float16 bbuf[2][2 * 32 * 16];  // 2 x 2KB TDM buffers

    int bid = blockIdx.x;
    int xt = bid % XT;
    int y  = (bid / XT) % Hh;
    int b  = bid / (XT * Hh);
    int lane = threadIdx.x;
    int m = lane & 15;

    // ---- stage the 3x18x256 input patch (zero padded) as f16, [row][x][cin]
    for (int it = lane; it < 3 * 256; it += 32) {
        int yyi = it >> 8;
        int cin = it & 255;
        int yy = y + yyi - 1;
        const float* src = (cin < Cch) ? low : high;
        int cc = (cin < Cch) ? cin : cin - Cch;
        bool yok = (yy >= 0) && (yy < Hh);
        const float* rowp = &src[((size_t)(b * Cch + cc) * Hh + (yok ? yy : 0)) * Wd];
#pragma unroll
        for (int xc = 0; xc < 18; ++xc) {
            int xx = xt * 16 + xc - 1;
            float v = (yok && xx >= 0 && xx < Wd) ? rowp[xx] : 0.0f;
            patch[((size_t)yyi * 18 + xc) * PS + cin] = (_Float16)v;
        }
    }
    __syncthreads();

    v8f c0 = (v8f)0.0f, c1 = (v8f)0.0f;
    int laneoff = (lane >= 16) ? 8 : 0;

#if HAVE_TDM
    unsigned ldsb[2];
    ldsb[0] = (unsigned)(uintptr_t)&bbuf[0][0];
    ldsb[1] = (unsigned)(uintptr_t)&bbuf[1][0];
    tdm_load_1d(ldsb[0], wB, 512);                  // 2KB B block for kb=0
#endif

    for (int kb = 0; kb < KOFF / 32; ++kb) {
#if HAVE_TDM
        if (kb + 1 < KOFF / 32) {
            tdm_load_1d(ldsb[(kb + 1) & 1], wB + (size_t)(kb + 1) * 1024, 512);
            __builtin_amdgcn_s_wait_tensorcnt(1);
        } else {
            __builtin_amdgcn_s_wait_tensorcnt(0);
        }
        const _Float16* bbase = &bbuf[kb & 1][0];
#else
        const _Float16* bbase = &wB[(size_t)kb * 1024];
#endif
        int tap = kb >> 3;                 // uniform (SALU)
        int cinb = (kb & 7) * 32 + laneoff;
        int ty = tap / 3, tx = tap % 3;    // uniform
        v16h a = load_a_frag(&patch[((size_t)(ty * 18 + m + tx)) * PS + cinb]);
        v16h b0 = *(const v16h*)&bbase[((size_t)(0 * 32 + lane)) * 16];
        v16h b1 = *(const v16h*)&bbase[((size_t)(1 * 32 + lane)) * 16];
        c0 = __builtin_amdgcn_wmma_f32_16x16x32_f16(false, a, false, b0, (short)0, c0, false, false);
        c1 = __builtin_amdgcn_wmma_f32_16x16x32_f16(false, a, false, b1, (short)0, c1, false, false);
    }

    // ---- vectorized store: each lane owns 8 contiguous x for one channel ----
    int hi8 = (lane >> 4) << 3;
    int xo  = xt * 16 + hi8;
    {
        int n0 = lane & 15;                     // always < 18
        float bias = b_off[n0];
        v4f lo, hi;
#pragma unroll
        for (int r = 0; r < 4; ++r) { lo[r] = c0[r] + bias; hi[r] = c0[r + 4] + bias; }
        float* dst = &offs[((size_t)(b * 18 + n0) * Hh + y) * Wd + xo];
        *(v4f*)dst = lo;
        *(v4f*)(dst + 4) = hi;
    }
    {
        int n1 = 16 + (lane & 15);
        if (n1 < 18) {
            float bias = b_off[n1];
            v4f lo, hi;
#pragma unroll
            for (int r = 0; r < 4; ++r) { lo[r] = c1[r] + bias; hi[r] = c1[r + 4] + bias; }
            float* dst = &offs[((size_t)(b * 18 + n1) * Hh + y) * Wd + xo];
            *(v4f*)dst = lo;
            *(v4f*)(dst + 4) = hi;
        }
    }
}

// ---------------------------------------------------------------------------
// Fused gate(1x1 sigmoid) + deformable conv + residual. One wave per tile.
// ---------------------------------------------------------------------------
__global__ __launch_bounds__(32) void deform_fuse_kernel(
    const float* __restrict__ low, const float* __restrict__ high,
    const _Float16* __restrict__ wModB, const _Float16* __restrict__ wDefB,
    const float* __restrict__ b_def, const float* __restrict__ b_mod,
    const float* __restrict__ offs, float* __restrict__ out)
{
    constexpr int SS = 136;   // sampled-cin stride (halves)
    constexpr int CS = 264;   // cat-cin stride (halves)
    constexpr int GS = 20;    // gate row stride (floats), padded
    __shared__ __align__(16) _Float16 samp[9 * 16 * SS];    // bilinear samples [tap][m][cin]
    __shared__ __align__(16) _Float16 catpix[16 * CS];      // cat pixels [m][cin]
    __shared__ __align__(16) _Float16 bbuf[2][8 * 32 * 16]; // TDM double buffer (2 x 8KB)
    __shared__ __align__(16) float gate_s[128][GS];         // gate, [n][m] transposed
    __shared__ int   si[16][9][2];
    __shared__ float sw[16][9][4];

    int bid = blockIdx.x;
    int xt = bid % XT;
    int y  = (bid / XT) % Hh;
    int b  = bid / (XT * Hh);
    int lane = threadIdx.x;
    int mrow = lane & 15;
    int laneoff = (lane >= 16) ? 8 : 0;
    int hi8 = (lane >> 4) << 3;

    __builtin_prefetch(&high[((size_t)(b * Cch) * Hh + y) * Wd + xt * 16], 0, 1);

#if HAVE_TDM
    unsigned ldsb[2];
    ldsb[0] = (unsigned)(uintptr_t)&bbuf[0][0];
    ldsb[1] = (unsigned)(uintptr_t)&bbuf[1][0];
#endif

    // ---- stage 0a: bilinear metadata per (pixel, tap) ----
    for (int i = lane; i < 16 * 9; i += 32) {
        int m = i / 9, t = i - (i / 9) * 9;
        int x = xt * 16 + m;
        float dy = offs[((size_t)(b * 18 + 2 * t + 0) * Hh + y) * Wd + x];
        float dx = offs[((size_t)(b * 18 + 2 * t + 1) * Hh + y) * Wd + x];
        float py = (float)y + (float)(t / 3 - 1) + dy;
        float px = (float)x + (float)(t % 3 - 1) + dx;
        float fy = floorf(py), fx = floorf(px);
        float wy = py - fy,   wx = px - fx;
        si[m][t][0] = (int)fy;
        si[m][t][1] = (int)fx;
        sw[m][t][0] = (1.0f - wy) * (1.0f - wx);
        sw[m][t][1] = (1.0f - wy) * wx;
        sw[m][t][2] = wy * (1.0f - wx);
        sw[m][t][3] = wy * wx;
    }

    // ---- stage 0b: stage cat pixels [m][cin] (coalesced over m) ----
    for (int idx = lane; idx < 16 * 256; idx += 32) {
        int m   = idx & 15;
        int cin = idx >> 4;
        const float* src = (cin < Cch) ? low : high;
        int cc = (cin < Cch) ? cin : cin - Cch;
        float v = src[((size_t)(b * Cch + cc) * Hh + y) * Wd + xt * 16 + m];
        catpix[(size_t)m * CS + cin] = (_Float16)v;
    }
    __syncthreads();

    // ---- stage 0c: bilinear-sample low once per (tap, pixel, cin), branchless
    {
        int cstart = lane >> 4;   // 0 or 1 (cin parity), pixel = mrow
        const float* base = &low[(size_t)b * Cch * HW];
        for (int t = 0; t < 9; ++t) {
            int iy = si[mrow][t][0];
            int ix = si[mrow][t][1];
            bool y0ok = (iy >= 0) && (iy < Hh);
            bool y1ok = (iy + 1 >= 0) && (iy + 1 < Hh);
            bool x0ok = (ix >= 0) && (ix < Wd);
            bool x1ok = (ix + 1 >= 0) && (ix + 1 < Wd);
            float w00 = (y0ok && x0ok) ? sw[mrow][t][0] : 0.0f;
            float w01 = (y0ok && x1ok) ? sw[mrow][t][1] : 0.0f;
            float w10 = (y1ok && x0ok) ? sw[mrow][t][2] : 0.0f;
            float w11 = (y1ok && x1ok) ? sw[mrow][t][3] : 0.0f;
            int cy0 = min(max(iy, 0), Hh - 1);
            int cy1 = min(max(iy + 1, 0), Hh - 1);
            int cx0 = min(max(ix, 0), Wd - 1);
            int cx1 = min(max(ix + 1, 0), Wd - 1);
            int o00 = cy0 * Wd + cx0, o01 = cy0 * Wd + cx1;
            int o10 = cy1 * Wd + cx0, o11 = cy1 * Wd + cx1;
            for (int c = cstart; c < Cch; c += 2) {
                const float* bc = base + (size_t)c * HW;
                float v = w00 * bc[o00] + w01 * bc[o01] + w10 * bc[o10] + w11 * bc[o11];
                samp[((size_t)(t * 16 + mrow)) * SS + c] = (_Float16)v;
            }
        }
    }
    __syncthreads();

    // ---- stage 1: gate GEMM (M=16, N=128, K=256), sigmoid -> LDS [n][m] ----
    {
        v8f g[8];
#pragma unroll
        for (int nt = 0; nt < 8; ++nt) g[nt] = (v8f)0.0f;

#if HAVE_TDM
        tdm_load_1d(ldsb[0], wModB, 2048);          // 8KB B block for kb=0
#endif
        for (int kb = 0; kb < KMOD / 32; ++kb) {
#if HAVE_TDM
            if (kb + 1 < KMOD / 32) {
                tdm_load_1d(ldsb[(kb + 1) & 1], wModB + (size_t)(kb + 1) * 4096, 2048);
                __builtin_amdgcn_s_wait_tensorcnt(1);
            } else {
                __builtin_amdgcn_s_wait_tensorcnt(0);
            }
            const _Float16* bbase = &bbuf[kb & 1][0];
#else
            const _Float16* bbase = &wModB[(size_t)kb * 4096];
#endif
            v16h a = load_a_frag(&catpix[(size_t)mrow * CS + kb * 32 + laneoff]);
#pragma unroll
            for (int nt = 0; nt < 8; ++nt) {
                v16h bt = *(const v16h*)&bbase[((size_t)(nt * 32 + lane)) * 16];
                g[nt] = __builtin_amdgcn_wmma_f32_16x16x32_f16(false, a, false, bt, (short)0, g[nt], false, false);
            }
        }
#pragma unroll
        for (int nt = 0; nt < 8; ++nt) {
            int n = nt * 16 + (lane & 15);
            float bias = b_mod[n];
            v4f glo, ghi;
#pragma unroll
            for (int r = 0; r < 4; ++r) {
                glo[r] = 1.0f / (1.0f + __expf(-(g[nt][r] + bias)));
                ghi[r] = 1.0f / (1.0f + __expf(-(g[nt][r + 4] + bias)));
            }
            *(v4f*)&gate_s[n][hi8] = glo;
            *(v4f*)&gate_s[n][hi8 + 4] = ghi;
        }
    }
    __syncthreads();

    // ---- stage 2: deformable GEMM (M=16, N=128, K=1152 tap-major) ----
    v8f d[8];
#pragma unroll
    for (int nt = 0; nt < 8; ++nt) d[nt] = (v8f)0.0f;

#if HAVE_TDM
    tdm_load_1d(ldsb[0], wDefB, 2048);              // 8KB B block for kb=0
#endif
    for (int kb = 0; kb < KDEF / 32; ++kb) {
#if HAVE_TDM
        if (kb + 1 < KDEF / 32) {
            tdm_load_1d(ldsb[(kb + 1) & 1], wDefB + (size_t)(kb + 1) * 4096, 2048);
            __builtin_amdgcn_s_wait_tensorcnt(1);
        } else {
            __builtin_amdgcn_s_wait_tensorcnt(0);
        }
        const _Float16* bbase = &bbuf[kb & 1][0];
#else
        const _Float16* bbase = &wDefB[(size_t)kb * 4096];
#endif
        int tap  = kb >> 2;                          // uniform
        int cinb = (kb & 3) * 32 + laneoff;
        v16h a = load_a_frag(&samp[((size_t)(tap * 16 + mrow)) * SS + cinb]);
#pragma unroll
        for (int nt = 0; nt < 8; ++nt) {
            v16h bt = *(const v16h*)&bbase[((size_t)(nt * 32 + lane)) * 16];
            d[nt] = __builtin_amdgcn_wmma_f32_16x16x32_f16(false, a, false, bt, (short)0, d[nt], false, false);
        }
    }
    __syncthreads();

    // ---- stage 3: epilogue  out = (def + b_def) * gate + high  (all b128) ----
#pragma unroll
    for (int nt = 0; nt < 8; ++nt) {
        int n = nt * 16 + (lane & 15);
        float bias = b_def[n];
        v4f gl = *(const v4f*)&gate_s[n][hi8];
        v4f gh = *(const v4f*)&gate_s[n][hi8 + 4];
        size_t idx = ((size_t)(b * Cch + n) * Hh + y) * Wd + xt * 16 + hi8;
        v4f h0 = *(const v4f*)&high[idx];
        v4f h1 = *(const v4f*)&high[idx + 4];
        v4f o0, o1;
#pragma unroll
        for (int r = 0; r < 4; ++r) {
            o0[r] = (d[nt][r] + bias) * gl[r] + h0[r];
            o1[r] = (d[nt][r + 4] + bias) * gh[r] + h1[r];
        }
        *(v4f*)&out[idx] = o0;
        *(v4f*)&out[idx + 4] = o1;
    }
}

// ---------------------------------------------------------------------------
extern "C" void kernel_launch(void* const* d_in, const int* in_sizes, int n_in,
                              void* d_out, int out_size, void* d_ws, size_t ws_size,
                              hipStream_t stream)
{
    (void)in_sizes; (void)n_in; (void)out_size; (void)ws_size;

    const float* low   = (const float*)d_in[0];
    const float* high  = (const float*)d_in[1];
    const float* w_off = (const float*)d_in[2];
    const float* b_off = (const float*)d_in[3];
    const float* w_def = (const float*)d_in[4];
    const float* b_def = (const float*)d_in[5];
    const float* w_mod = (const float*)d_in[6];
    const float* b_mod = (const float*)d_in[7];
    float* out = (float*)d_out;

    char* ws = (char*)d_ws;
    _Float16* wOffB = (_Float16*)ws;
    _Float16* wModB = wOffB + WOFF_HALVES;
    _Float16* wDefB = wModB + WMOD_HALVES;
    float*    offs  = (float*)(ws + (size_t)(WOFF_HALVES + WMOD_HALVES + WDEF_HALVES) * 2);

    // k = tap*cins + cin ; src index = n*Kdim + cin*taps + tap
    pack_b_kernel<<<(WOFF_HALVES + 255) / 256, 256, 0, stream>>>(w_off, wOffB, KOFF, 2, 18, 9, 8);
    pack_b_kernel<<<(WMOD_HALVES + 255) / 256, 256, 0, stream>>>(w_mod, wModB, KMOD, 8, 128, 1, 8);
    pack_b_kernel<<<(WDEF_HALVES + 255) / 256, 256, 0, stream>>>(w_def, wDefB, KDEF, 8, 128, 9, 7);

    dim3 grid(Bn * Hh * XT);   // 4096 tiles, one wave each
    offset_conv_kernel<<<grid, 32, 0, stream>>>(low, high, wOffB, b_off, offs);
    deform_fuse_kernel<<<grid, 32, 0, stream>>>(low, high, wModB, wDefB, b_def, b_mod, offs, out);
}
